// Gnn_16561393893503
// MI455X (gfx1250) — matrix operand order repaired and checked by reading, not compiled, not run on hardware
//
#include <hip/hip_runtime.h>
#include <hip/hip_bf16.h>

// ---------------- types ----------------
typedef __attribute__((ext_vector_type(16))) __bf16 bf16x16;
typedef __attribute__((ext_vector_type(8)))  float  f32x8;
typedef unsigned int u32x4 __attribute__((ext_vector_type(4)));
typedef unsigned short u16;

#define D1 768
#define D3 2304
#define D4 3072
#define NCLS 16

// ---------------- helpers ----------------
__device__ __forceinline__ u16 f2bf(float f) {
    unsigned u = __float_as_uint(f);
    unsigned r = (u + 0x7fffu + ((u >> 16) & 1u)) >> 16;   // RNE
    return (u16)r;
}
__device__ __forceinline__ float bf2f(u16 h) {
    return __uint_as_float(((unsigned)h) << 16);
}
__device__ __forceinline__ void atomAddF(float* p, float v) {
#if defined(__HIP_PLATFORM_AMD__)
    unsafeAtomicAdd(p, v);          // native global_atomic_add_f32
#else
    atomicAdd(p, v);
#endif
}
__device__ __forceinline__ int fkey(float f) {   // order-preserving float->int
    int i = __float_as_int(f);
    return i >= 0 ? i : (i ^ 0x7fffffff);
}
__device__ __forceinline__ float fdecode(int k) {
    int i = k >= 0 ? k : (k ^ 0x7fffffff);
    return __int_as_float(i);
}

// ---------------- init kernels ----------------
__global__ void fill_f32(float* p, float v, long n) {
    long i = (long)blockIdx.x * blockDim.x + threadIdx.x;
    long st = (long)gridDim.x * blockDim.x;
    for (; i < n; i += st) p[i] = v;
}
__global__ void fill_i32(int* p, int v, long n) {
    long i = (long)blockIdx.x * blockDim.x + threadIdx.x;
    long st = (long)gridDim.x * blockDim.x;
    for (; i < n; i += st) p[i] = v;
}
__global__ void bias_bcast(float* out, const float* b, long rows, int dim) {
    long n = rows * dim;
    long i = (long)blockIdx.x * blockDim.x + threadIdx.x;
    long st = (long)gridDim.x * blockDim.x;
    for (; i < n; i += st) out[i] = b[i % dim];
}

// ---------------- batchnorm ----------------
__global__ void bn_sums(const float* x, float* colsum, float* colsq, int n) {
    int c = blockIdx.x * 256 + threadIdx.x;   // gridDim.x = D3/256
    float s = 0.f, q = 0.f;
    for (int r = blockIdx.y; r < n; r += gridDim.y) {
        float v = x[(size_t)r * D3 + c];
        s += v; q += v * v;
    }
    atomAddF(colsum + c, s);
    atomAddF(colsq + c, q);
}
__global__ void bn_finalize(const float* colsum, const float* colsq,
                            const float* gamma, const float* beta,
                            float* scale, float* shift, int n) {
    int c = blockIdx.x * 256 + threadIdx.x;
    if (c >= D3) return;
    float mean = colsum[c] / (float)n;
    float var  = colsq[c] / (float)n - mean * mean;
    float inv  = rsqrtf(var + 1e-5f);
    float sc   = gamma[c] * inv;
    scale[c] = sc;
    shift[c] = beta[c] - mean * sc;
}
__global__ void normalize_bf16(const float* x, const float* scale, const float* shift,
                               u16* xnb, long n) {
    long i = (long)blockIdx.x * blockDim.x + threadIdx.x;
    long st = (long)gridDim.x * blockDim.x;
    for (; i < n; i += st) {
        int c = (int)(i % D3);
        xnb[i] = f2bf(fmaf(x[i], scale[c], shift[c]));
    }
}
// Wt[n][k] = bf16(W[k][n])  (K-major weights; matches WMMA B-fragment layout)
__global__ void transpose_to_bf16(const float* W, u16* Wt, int K, int Nc) {
    long n = (long)K * Nc;
    long i = (long)blockIdx.x * blockDim.x + threadIdx.x;
    long st = (long)gridDim.x * blockDim.x;
    for (; i < n; i += st) {
        int nc = (int)(i / K);
        int k  = (int)(i % K);
        Wt[i] = f2bf(W[(size_t)k * Nc + nc]);
    }
}

// ---------------- WMMA bf16 GEMM: C[M,Nc] = A[M,K] * Bt[Nc,K]^T ----------------
// A row-major [M,K] bf16; Bt K-major [Nc,K] bf16; C row-major f32.
// grid = (M/16, Nc/256); block = 128 (4 waves, each owns a 16x64 strip).
// No LDS/barriers; fragments loaded straight from global (B is L2-resident).
// Software-pipelined: double-buffered fragment sets so loads of set S1 overlap
// the WMMAs of set S0 (distinct VGPRs -> staged s_wait_loadcnt instead of 0).
// Requires K % 64 == 0 and K >= 128 (K = 2304 / 3072 here).
union Frag { u32x4 u[2]; bf16x16 v; };

__global__ __launch_bounds__(128)
void gemm_bf16_tn(const u16* __restrict__ A, const u16* __restrict__ Bt,
                  float* __restrict__ C, int M, int K, int Nc) {
    const int lane = threadIdx.x & 31;
    const int wid  = threadIdx.x >> 5;
    const int m0 = blockIdx.x * 16;
    const int n0 = blockIdx.y * 256;
    const int r  = lane & 15;          // A row / B col within tile
    const int hh = lane >> 4;          // lane half selects K sub-range

    const u16* arow = A + (size_t)(m0 + r) * K;
    const u16* a0 = arow + 16 * hh;          // A chunk 0: K = k0+16h..+7
    const u16* a1 = arow + 16 * hh + 8;      // A chunk 1: K = k0+16h+8..+7
    // B fragment tile t: col = n0 + wid*64 + t*16 + r, K = k0 + 16*hh .. +15
    const u16* bbase = Bt + (size_t)(n0 + wid * 64 + r) * K + 16 * hh;
    const size_t bstep = (size_t)16 * K;     // next 16-wide N tile

    f32x8 acc[4] = {};
    Frag aS0, aS1, bS0[4], bS1[4];

    auto LD = [&](int k0, Frag& af, Frag (&bf)[4]) {
        af.u[0] = *(const u32x4*)(a0 + k0);
        af.u[1] = *(const u32x4*)(a1 + k0);
        #pragma unroll
        for (int t = 0; t < 4; ++t) {
            const u32x4* p = (const u32x4*)(bbase + (size_t)t * bstep + k0);
            bf[t].u[0] = p[0];
            bf[t].u[1] = p[1];
        }
    };
    auto MM = [&](Frag& af, Frag (&bf)[4]) {
        #pragma unroll
        for (int t = 0; t < 4; ++t)
            acc[t] = __builtin_amdgcn_wmma_f32_16x16x32_bf16(
                false, af.v, false, bf[t].v, (short)0, acc[t], false, false);
    };

    LD(0, aS0, bS0);
    int k0 = 0;
    for (; k0 + 64 < K; k0 += 64) {
        LD(k0 + 32, aS1, bS1);                       // prefetch next set
        __builtin_prefetch(a0 + k0 + 96, 0, 3);      // global_prefetch_b8
        MM(aS0, bS0);                                // overlap with S1 loads
        LD(k0 + 64, aS0, bS0);
        MM(aS1, bS1);
    }
    LD(k0 + 32, aS1, bS1);                           // last chunk
    MM(aS0, bS0);
    MM(aS1, bS1);

    // epilogue: VGPR d -> row m0 + d + 8*hh, col = r
    #pragma unroll
    for (int t = 0; t < 4; ++t) {
        int c = n0 + wid * 64 + t * 16 + r;
        int rbase = m0 + 8 * hh;
        #pragma unroll
        for (int d = 0; d < 8; ++d)
            C[(size_t)(rbase + d) * Nc + c] = acc[t][d];
    }
}

// ---------------- attention dots: as[i]=h[i].a_src, ad[i]=h[i].a_dst ----------------
__global__ void dot_av(const float* __restrict__ h, const float* __restrict__ asrc,
                       const float* __restrict__ adst, float* as_, float* ad_) {
    int i = blockIdx.x;
    const float* row = h + (size_t)i * D1;
    float s = 0.f, d = 0.f;
    for (int c = threadIdx.x; c < D1; c += 256) {
        float v = row[c];
        s += v * asrc[c];
        d += v * adst[c];
    }
    __shared__ float ss[256], sd[256];
    ss[threadIdx.x] = s; sd[threadIdx.x] = d;
    __syncthreads();
    for (int off = 128; off > 0; off >>= 1) {
        if (threadIdx.x < off) {
            ss[threadIdx.x] += ss[threadIdx.x + off];
            sd[threadIdx.x] += sd[threadIdx.x + off];
        }
        __syncthreads();
    }
    if (threadIdx.x == 0) { as_[i] = ss[0]; ad_[i] = sd[0]; }
}

// ---------------- edge softmax passes ----------------
__global__ void edge_logits(const int* ei, const float* as_, const float* ad_,
                            float* eraw, int* segmax, int E_, int N_) {
    int e = blockIdx.x * blockDim.x + threadIdx.x;
    int EE = E_ + N_;
    if (e >= EE) return;
    int s = (e < E_) ? ei[e]       : (e - E_);
    int d = (e < E_) ? ei[E_ + e]  : (e - E_);
    float v = as_[s] + ad_[d];
    v = v > 0.f ? v : 0.2f * v;                 // leaky_relu
    eraw[e] = v;
    atomicMax(segmax + d, fkey(v));
}
__global__ void edge_exp(const int* ei, float* eraw, const int* segmax,
                         float* segsum, int E_, int N_) {
    int e = blockIdx.x * blockDim.x + threadIdx.x;
    int EE = E_ + N_;
    if (e >= EE) return;
    int d = (e < E_) ? ei[E_ + e] : (e - E_);
    float ex = __expf(eraw[e] - fdecode(segmax[d]));
    eraw[e] = ex;                               // becomes exp-numerator
    atomAddF(segsum + d, ex);
}
__global__ void scatter_edges(const int* ei, const float* eexp, const float* segsum,
                              const float* __restrict__ h, float* out, int E_, int N_) {
    int e = blockIdx.x;                         // grid = E+N
    int s = (e < E_) ? ei[e]      : (e - E_);
    int d = (e < E_) ? ei[E_ + e] : (e - E_);
    float coef = eexp[e] / segsum[d];
    const float* hs = h + (size_t)s * D1;
    float* od = out + (size_t)d * D1;
    __builtin_prefetch(hs + threadIdx.x, 0, 3);
    for (int c = threadIdx.x; c < D1; c += blockDim.x)
        atomAddF(od + c, coef * hs[c]);
}

// ---------------- concat [out1 | xn] -> bf16 ----------------
__global__ void concat_bf16(const float* out1, const u16* xnb, u16* cat, long rows) {
    long n = rows * D4;
    long i = (long)blockIdx.x * blockDim.x + threadIdx.x;
    long st = (long)gridDim.x * blockDim.x;
    for (; i < n; i += st) {
        long row = i / D4; int c = (int)(i % D4);
        cat[i] = (c < D1) ? f2bf(out1[row * D1 + c]) : xnb[row * D3 + (c - D1)];
    }
}

// ---------------- skinny classifiers (C=16) ----------------
__global__ void head_bf16(const u16* A, const float* W, const float* b,
                          float* out, int K) {      // grid=N, block=64
    int i = blockIdx.x;
    int c = threadIdx.x & 15, g = threadIdx.x >> 4;  // 4 k-groups
    const u16* row = A + (size_t)i * K;
    float sum = 0.f;
    for (int k = g; k < K; k += 4) sum += bf2f(row[k]) * W[k * NCLS + c];
    __shared__ float red[64];
    red[threadIdx.x] = sum;
    __syncthreads();
    if (g == 0)
        out[(size_t)i * NCLS + c] = red[c] + red[16 + c] + red[32 + c] + red[48 + c] + b[c];
}
__global__ void head_f32(const float* A, const float* W, const float* b,
                         float* out, int K) {       // grid=N, block=64
    int i = blockIdx.x;
    int c = threadIdx.x & 15, g = threadIdx.x >> 4;
    const float* row = A + (size_t)i * K;
    float sum = 0.f;
    for (int k = g; k < K; k += 4) sum += row[k] * W[k * NCLS + c];
    __shared__ float red[64];
    red[threadIdx.x] = sum;
    __syncthreads();
    if (g == 0)
        out[(size_t)i * NCLS + c] = red[c] + red[16 + c] + red[32 + c] + red[48 + c] + b[c];
}

// ---------------- masked CE + accuracy ----------------
__global__ void loss_kernel(const float* pooler, const float* direct,
                            const int* target, float* accum, int n) {
    int i = blockIdx.x * blockDim.x + threadIdx.x;
    if (i >= n) return;
    int t = target[i];
    if (t < 0) return;
    const float* p = pooler + (size_t)i * NCLS;
    const float* q = direct + (size_t)i * NCLS;
    float mp = p[0], mq = q[0];
    int am = 0; float bv = p[0];
    #pragma unroll
    for (int j = 1; j < NCLS; ++j) {
        mp = fmaxf(mp, p[j]); mq = fmaxf(mq, q[j]);
        if (p[j] > bv) { bv = p[j]; am = j; }
    }
    float sp = 0.f, sq = 0.f;
    #pragma unroll
    for (int j = 0; j < NCLS; ++j) { sp += __expf(p[j] - mp); sq += __expf(q[j] - mq); }
    float nllp = (mp + __logf(sp)) - p[t];
    float nllq = (mq + __logf(sq)) - q[t];
    atomAddF(&accum[0], nllp);
    atomAddF(&accum[1], nllq);
    atomAddF(&accum[2], (am == t) ? 1.f : 0.f);
    atomAddF(&accum[3], 1.f);
}
__global__ void finalize_kernel(const float* accum, float* loss_out, float* acc_out) {
    if (threadIdx.x == 0 && blockIdx.x == 0) {
        float nv = accum[3];
        loss_out[0] = (accum[0] + accum[1]) / nv;
        acc_out[0]  = accum[2] / nv;
    }
}

// ---------------- host launcher ----------------
extern "C" void kernel_launch(void* const* d_in, const int* in_sizes, int n_in,
                              void* d_out, int out_size, void* d_ws, size_t ws_size,
                              hipStream_t stream) {
    const float* x       = (const float*)d_in[0];
    const int*   ei      = (const int*)d_in[1];
    const int*   target  = (const int*)d_in[2];
    const float* gamma   = (const float*)d_in[3];
    const float* beta    = (const float*)d_in[4];
    const float* W1      = (const float*)d_in[5];
    const float* asrc1   = (const float*)d_in[6];
    const float* adst1   = (const float*)d_in[7];
    const float* b1      = (const float*)d_in[8];
    const float* W2      = (const float*)d_in[9];
    const float* asrc2   = (const float*)d_in[10];
    const float* adst2   = (const float*)d_in[11];
    const float* b2      = (const float*)d_in[12];
    const float* Wp      = (const float*)d_in[13];
    const float* bp      = (const float*)d_in[14];
    const float* Wd      = (const float*)d_in[15];
    const float* bd      = (const float*)d_in[16];

    const int N = in_sizes[2];          // 50000
    const int E = in_sizes[1] / 2;      // 400000
    const int EE = E + N;

    // workspace carve-out
    char* w = (char*)d_ws;
    auto carve = [&](size_t bytes) -> void* {
        void* p = (void*)w;
        w += (bytes + 255) & ~(size_t)255;
        return p;
    };
    float* colsum  = (float*)carve(D3 * 4);
    float* colsq   = (float*)carve(D3 * 4);
    float* scale   = (float*)carve(D3 * 4);
    float* shift   = (float*)carve(D3 * 4);
    float* accum   = (float*)carve(4 * 4);
    u16*   xnb     = (u16*)carve((size_t)N * D3 * 2);
    u16*   W1t     = (u16*)carve((size_t)D3 * D1 * 2);   // [D1][D3] K-major
    u16*   W2t     = (u16*)carve((size_t)D4 * D1 * 2);   // [D1][D4] K-major
    float* h       = (float*)carve((size_t)N * D1 * 4);  // reused layer1/layer2
    float* out1    = (float*)carve((size_t)N * D1 * 4);
    u16*   catb    = (u16*)carve((size_t)N * D4 * 2);
    float* as_     = (float*)carve((size_t)N * 4);
    float* ad_     = (float*)carve((size_t)N * 4);
    int*   segmax  = (int*)carve((size_t)N * 4);
    float* segsum  = (float*)carve((size_t)N * 4);
    float* eexp    = (float*)carve((size_t)EE * 4);
    float* direct  = (float*)carve((size_t)N * NCLS * 4);

    float* out_reg    = (float*)d_out;                    // [N, D1]
    float* pooler_reg = out_reg + (size_t)N * D1;         // [N, NCLS]
    float* loss_reg   = pooler_reg + (size_t)N * NCLS;    // [1]
    float* acc_reg    = loss_reg + 1;                     // [1]

    const int TB = 256;
    const int GS = 2048;                                  // grid-stride grid

    // --- BatchNorm ---
    fill_f32<<<4, TB, 0, stream>>>(colsum, 0.f, D3);
    fill_f32<<<4, TB, 0, stream>>>(colsq, 0.f, D3);
    fill_f32<<<1, 32, 0, stream>>>(accum, 0.f, 4);
    bn_sums<<<dim3(D3 / 256, 64), TB, 0, stream>>>(x, colsum, colsq, N);
    bn_finalize<<<D3 / 256, TB, 0, stream>>>(colsum, colsq, gamma, beta, scale, shift, N);
    normalize_bf16<<<GS, TB, 0, stream>>>(x, scale, shift, xnb, (long)N * D3);

    // --- weights: fp32 row-major -> bf16 K-major (transposed) ---
    transpose_to_bf16<<<GS, TB, 0, stream>>>(W1, W1t, D3, D1);
    transpose_to_bf16<<<GS, TB, 0, stream>>>(W2, W2t, D4, D1);

    // ================= GAT layer 1 =================
    gemm_bf16_tn<<<dim3(N / 16, D1 / 256), 128, 0, stream>>>(xnb, W1t, h, N, D3, D1);
    dot_av<<<N, TB, 0, stream>>>(h, asrc1, adst1, as_, ad_);
    fill_i32<<<GS, TB, 0, stream>>>(segmax, (int)0x80000000, N);
    fill_f32<<<GS, TB, 0, stream>>>(segsum, 0.f, N);
    bias_bcast<<<GS, TB, 0, stream>>>(out1, b1, N, D1);
    edge_logits<<<(EE + TB - 1) / TB, TB, 0, stream>>>(ei, as_, ad_, eexp, segmax, E, N);
    edge_exp<<<(EE + TB - 1) / TB, TB, 0, stream>>>(ei, eexp, segmax, segsum, E, N);
    scatter_edges<<<EE, TB, 0, stream>>>(ei, eexp, segsum, h, out1, E, N);

    // ================= GAT layer 2 =================
    concat_bf16<<<GS, TB, 0, stream>>>(out1, xnb, catb, N);
    gemm_bf16_tn<<<dim3(N / 16, D1 / 256), 128, 0, stream>>>(catb, W2t, h, N, D4, D1);
    dot_av<<<N, TB, 0, stream>>>(h, asrc2, adst2, as_, ad_);
    fill_i32<<<GS, TB, 0, stream>>>(segmax, (int)0x80000000, N);
    fill_f32<<<GS, TB, 0, stream>>>(segsum, 0.f, N);
    bias_bcast<<<GS, TB, 0, stream>>>(out_reg, b2, N, D1);
    edge_logits<<<(EE + TB - 1) / TB, TB, 0, stream>>>(ei, as_, ad_, eexp, segmax, E, N);
    edge_exp<<<(EE + TB - 1) / TB, TB, 0, stream>>>(ei, eexp, segmax, segsum, E, N);
    scatter_edges<<<EE, TB, 0, stream>>>(ei, eexp, segsum, h, out_reg, E, N);

    // ================= heads + loss =================
    head_bf16<<<N, 64, 0, stream>>>(xnb, Wd, bd, direct, D3);
    head_f32<<<N, 64, 0, stream>>>(out_reg, Wp, bp, pooler_reg, D1);
    loss_kernel<<<(N + TB - 1) / TB, TB, 0, stream>>>(pooler_reg, direct, target, accum, N);
    finalize_kernel<<<1, 32, 0, stream>>>(accum, loss_reg, acc_reg);
}
